// AGCN_TF_model_83150566851164
// MI455X (gfx1250) — matrix operand order repaired and checked by reading, not compiled, not run on hardware
//
#include <hip/hip_runtime.h>
#include <hip/hip_bf16.h>
#include <math.h>

// ---------------- dims ----------------
// BS=1024, P=128, S=128, HADJ=64, HG=128, NH=4, IND=40, N=168 (pad 192 for 32-row tiling)
static const int kBS = 1024;
static const int kNP = 192;   // padded sequence dim (168 -> 192)

typedef __attribute__((ext_vector_type(16))) _Float16 v16h;
typedef __attribute__((ext_vector_type(8)))  float    v8f;

union FragH { uint4 u[2]; v16h h; };

// ---------------- block reductions ----------------
__device__ __forceinline__ float bred_sum(float v, float* buf) {
  int t = threadIdx.x;
  buf[t] = v; __syncthreads();
  for (int o = blockDim.x >> 1; o > 0; o >>= 1) {
    if (t < o) buf[t] += buf[t + o];
    __syncthreads();
  }
  float r = buf[0]; __syncthreads(); return r;
}
__device__ __forceinline__ float bred_max(float v, float* buf) {
  int t = threadIdx.x;
  buf[t] = v; __syncthreads();
  for (int o = blockDim.x >> 1; o > 0; o >>= 1) {
    if (t < o) buf[t] = fmaxf(buf[t], buf[t + o]);
    __syncthreads();
  }
  float r = buf[0]; __syncthreads(); return r;
}
__device__ __forceinline__ float bred_min(float v, float* buf) {
  int t = threadIdx.x;
  buf[t] = v; __syncthreads();
  for (int o = blockDim.x >> 1; o > 0; o >>= 1) {
    if (t < o) buf[t] = fminf(buf[t], buf[t + o]);
    __syncthreads();
  }
  float r = buf[0]; __syncthreads(); return r;
}

// ---------------- weight prep: f32 -> f16, transposed (NxK), zero-padded ----------------
// packed layout in W (f16 elements):
//  [0      ) spa_w1t  (64,128)   8192
//  [8192   ) spa_w2t  (64,64)    4096   rows n>=40 zero
//  [12288  ) tem_w1t  (64,64)    4096   cols k>=40 zero
//  [16384  ) tem_w2t  (128,64)   8192
//  [24576  ) sgnn_wt  (128,128) 16384
//  [40960  ) tgnn_wt  (128,64)   8192   cols k>=40 zero
//  [49152  ) q_wt     (4,128,128) 65536
//  [114688 ) k_wt     (4,128,128) 65536
//  [180224 ) v_wt     (4,128,128) 65536
// total 245760 f16
__global__ void prep_weights_kernel(const float* __restrict__ spa_w1, const float* __restrict__ spa_w2,
                                    const float* __restrict__ tem_w1, const float* __restrict__ tem_w2,
                                    const float* __restrict__ sgnn_w, const float* __restrict__ tgnn_w,
                                    const float* __restrict__ q_w, const float* __restrict__ k_w,
                                    const float* __restrict__ v_w, _Float16* __restrict__ W)
{
  for (int i = blockIdx.x * blockDim.x + threadIdx.x; i < 245760; i += gridDim.x * blockDim.x) {
    float v = 0.f;
    if (i < 8192)        { int n = i >> 7, k = i & 127;                 v = spa_w1[k * 64 + n]; }
    else if (i < 12288)  { int l = i - 8192;  int n = l >> 6, k = l & 63; v = (n < 40) ? spa_w2[k * 40 + n] : 0.f; }
    else if (i < 16384)  { int l = i - 12288; int n = l >> 6, k = l & 63; v = (k < 40) ? tem_w1[k * 64 + n] : 0.f; }
    else if (i < 24576)  { int l = i - 16384; int n = l >> 6, k = l & 63; v = tem_w2[k * 128 + n]; }
    else if (i < 40960)  { int l = i - 24576; int n = l >> 7, k = l & 127; v = sgnn_w[k * 128 + n]; }
    else if (i < 49152)  { int l = i - 40960; int n = l >> 6, k = l & 63; v = (k < 40) ? tgnn_w[k * 128 + n] : 0.f; }
    else if (i < 114688) { int l = i - 49152;  int h = l >> 14, r = l & 16383; int n = r >> 7, k = r & 127; v = q_w[h * 16384 + k * 128 + n]; }
    else if (i < 180224) { int l = i - 114688; int h = l >> 14, r = l & 16383; int n = r >> 7, k = r & 127; v = k_w[h * 16384 + k * 128 + n]; }
    else                 { int l = i - 180224; int h = l >> 14, r = l & 16383; int n = r >> 7, k = r & 127; v = v_w[h * 16384 + k * 128 + n]; }
    W[i] = (_Float16)v;
  }
}

// ---------------- feature extraction: one block per signal row (128 samples) -------------
__global__ __launch_bounds__(128) void feat_kernel(const float* __restrict__ X,
                                                   float* __restrict__ feats)
{
  const int row = blockIdx.x;           // 0 .. bs*128-1
  const int t = threadIdx.x;            // 0..127
  __shared__ float s_sig[128];
  __shared__ float s_red[128];
  __shared__ float s_cos[128];
  __shared__ float s_sin[128];
  __shared__ float s_psd[128];
  __shared__ float s_med;

  const float TWO_PI = 6.283185307179586f;
  float x = X[(long)row * 128 + t];
  s_sig[t] = x;
  float ang = TWO_PI * (float)t * (1.0f / 128.0f);
  s_cos[t] = __cosf(ang);
  s_sin[t] = __sinf(ang);
  __syncthreads();

  // ---- temporal ----
  float sum   = bred_sum(x, s_red);
  float sumsq = bred_sum(x * x, s_red);
  float mx    = bred_max(x, s_red);
  float mn    = bred_min(x, s_red);
  float mean  = sum * (1.0f / 128.0f);
  float var   = (sumsq - 128.0f * mean * mean) * (1.0f / 127.0f);
  float stdv  = sqrtf(var);
  float rms   = sqrtf(sumsq * (1.0f / 128.0f));
  float ptp   = mx - mn;

  float e  = __expf(x - mx);
  float se = bred_sum(e, s_red);
  float sxe = bred_sum(x * e, s_red);
  float ent = (mx + __logf(se)) - sxe / se;

  float xc = fminf(fmaxf(x, -1.0f + 1e-7f), 1.0f - 1e-7f);
  float a1 = asinf(xc);
  float a2 = atanf(x);
  float s1 = bred_sum(a1, s_red);
  float q1 = bred_sum(a1 * a1, s_red);
  float std_asin = sqrtf((q1 - s1 * s1 * (1.0f / 128.0f)) * (1.0f / 127.0f));
  float s2 = bred_sum(a2, s_red);
  float q2 = bred_sum(a2 * a2, s_red);
  float std_atan = sqrtf((q2 - s2 * s2 * (1.0f / 128.0f)) * (1.0f / 127.0f));

  float c  = x - mean;
  float c2 = c * c;
  float m3 = bred_sum(c2 * c, s_red) * (1.0f / 128.0f);
  float m4 = bred_sum(c2 * c2, s_red) * (1.0f / 128.0f);
  float kurt = m4 / (var * var) - 3.0f;
  float skew = m3 / (var * stdv);

  // ---- frequency: 128-pt DFT, bin k = t ----
  float re = 0.f, im = 0.f;
  for (int n = 0; n < 128; n++) {
    int idx = (t * n) & 127;
    float sv = s_sig[n];
    re += sv * s_cos[idx];
    im -= sv * s_sin[idx];
  }
  float amp = sqrtf(re * re + im * im);
  float psd = (re * re + im * im) * (1.0f / 128.0f);
  s_psd[t] = psd;
  float freq = ((t < 64) ? (float)t : (float)(t - 128)) * (1.0f / 128.0f);
  __syncthreads();

  float psd_sum   = bred_sum(psd, s_red);
  float mean_freq = bred_sum(freq * psd, s_red) / psd_sum;
  float occ       = bred_sum((freq < 0.5f) ? psd : 0.f, s_red) / psd_sum;
  float pbw       = sqrtf(bred_sum(psd * psd, s_red) / psd_sum);
  float max_psd   = bred_max(psd, s_red);
  float max_amp   = bred_max(amp, s_red);
  // argmax of |fv| (first index of max)
  float idxf = (amp == max_amp) ? (float)t : 1e9f;
  float amin = bred_min(idxf, s_red);
  int kmax = (int)amin;
  float f_max = ((kmax < 64) ? (float)kmax : (float)(kmax - 128)) * (1.0f / 128.0f);
  // median_freq: stable-ascending-sort rank == 64
  int rank = 0;
  float myp = s_psd[t];
  for (int j = 0; j < 128; j++) {
    float pj = s_psd[j];
    rank += (pj < myp) || (pj == myp && j < t);
  }
  if (rank == 64) s_med = freq;
  __syncthreads();
  float median_freq = s_med;

  if (t == 0) {
    float* o = feats + (long)row * 20;
    o[0] = mx;  o[1] = mn;  o[2] = stdv; o[3] = rms; o[4] = mean;
    o[5] = ptp; o[6] = var; o[7] = ent;  o[8] = std_asin; o[9] = std_atan;
    o[10] = kurt; o[11] = skew;
    o[12] = mean_freq; o[13] = median_freq; o[14] = psd_sum; o[15] = occ;
    o[16] = pbw; o[17] = max_psd; o[18] = max_amp; o[19] = f_max;
  }
}

// ---------------- cumsum + normalize, emit Xf (128x64 f16) and Xt (64x128 f16) -----------
__global__ __launch_bounds__(128) void cumsum_norm_kernel(const float* __restrict__ feats,
                                                          _Float16* __restrict__ Xf_b,
                                                          _Float16* __restrict__ Xt_b)
{
  int b = blockIdx.x;
  int t = threadIdx.x;
  __shared__ float F[128][40];
  __shared__ float red[128];
  for (int i = t; i < 128 * 20; i += 128) {
    int p = i / 20, f = i % 20;
    F[p][f] = feats[((long)b * 128 + p) * 20 + f];
  }
  __syncthreads();
  if (t < 20) {
    float cacc = 0.f;
    for (int p = 0; p < 128; p++) {
      cacc += F[p][t];
      float a = fmaxf(fabsf(cacc), 1e-12f);
      F[p][20 + t] = cacc / sqrtf(a);
    }
  }
  __syncthreads();
  float ss = 0.f;
  for (int i = t; i < 128 * 40; i += 128) {
    float v = F[i / 40][i % 40];
    ss += v * v;
  }
  float inv = 1.0f / sqrtf(bred_sum(ss, red));
  for (int i = t; i < 128 * 64; i += 128) {
    int p = i >> 6, f = i & 63;
    float v = (f < 40) ? F[p][f] * inv : 0.f;
    Xf_b[((long)b * 128 + p) * 64 + f] = (_Float16)v;
    Xt_b[((long)b * 64 + f) * 128 + p] = (_Float16)v;
  }
}

// ---------------- generic WMMA GEMM: C = epi(A @ Bt^T + bias) ----------------------------
// A: (M,K) f16 row-major, row stride == K.  Bt: (N,K) f16 row-major.  K multiple of 32.
// Register blocking: each wave computes a (16*TM) x (16*TN) output macro-tile; A fragments
// are reused across TN WMMAs, B fragments across TM WMMAs (TM*TN wmma per 2*(TM+TN) b128).
// EPI: 0 none, 1 tanh, 2 leaky_relu(0.01)
template <int EPI, int KDIM, int TM, int TN>
__global__ __launch_bounds__(128) void gemm_kernel(
    const _Float16* __restrict__ A, long lA,
    const _Float16* __restrict__ Bt, long lB,
    const float* __restrict__ bias, int biasN,
    _Float16* __restrict__ Ch, float* __restrict__ Cf, long lC,
    int M, int N, int ldc, int mOff, int mLim, int transposed)
{
  int b    = blockIdx.x;
  int wave = threadIdx.x >> 5;
  int lane = threadIdx.x & 31;
  int macN = N / (16 * TN);
  int mac  = blockIdx.y * 4 + wave;
  int totalMac = (M / (16 * TM)) * macN;
  if (mac >= totalMac) return;
  int tm = mac / macN, tn = mac % macN;
  int m16 = lane & 15, kh = lane >> 4;

  const _Float16* aRow[TM];
  const _Float16* bRow[TN];
#pragma unroll
  for (int i = 0; i < TM; i++)
    aRow[i] = A + (long)b * lA + (long)(tm * 16 * TM + i * 16 + m16) * KDIM;
#pragma unroll
  for (int j = 0; j < TN; j++)
    bRow[j] = Bt + (long)b * lB + (long)(tn * 16 * TN + j * 16 + m16) * KDIM;

  v8f acc[TM][TN];
#pragma unroll
  for (int i = 0; i < TM; i++)
#pragma unroll
    for (int j = 0; j < TN; j++)
      acc[i][j] = (v8f){0.f, 0.f, 0.f, 0.f, 0.f, 0.f, 0.f, 0.f};

#pragma unroll
  for (int k0 = 0; k0 < KDIM; k0 += 32) {
    FragH af[TM], bf[TN];
#pragma unroll
    for (int i = 0; i < TM; i++) {
      af[i].u[0] = *(const uint4*)(aRow[i] + k0 + kh * 8);
      af[i].u[1] = *(const uint4*)(aRow[i] + k0 + 16 + kh * 8);
    }
#pragma unroll
    for (int j = 0; j < TN; j++) {
      bf[j].u[0] = *(const uint4*)(bRow[j] + k0 + kh * 8);
      bf[j].u[1] = *(const uint4*)(bRow[j] + k0 + 16 + kh * 8);
    }
#pragma unroll
    for (int i = 0; i < TM; i++)
#pragma unroll
      for (int j = 0; j < TN; j++)
        acc[i][j] = __builtin_amdgcn_wmma_f32_16x16x32_f16(false, af[i].h, false, bf[j].h,
                                                           (short)0, acc[i][j], false, false);
  }

  float bv[TN];
#pragma unroll
  for (int j = 0; j < TN; j++) {
    int n = tn * 16 * TN + j * 16 + m16;
    bv[j] = (bias && n < biasN) ? bias[n] : 0.f;
  }
#pragma unroll
  for (int i = 0; i < TM; i++) {
#pragma unroll
    for (int j = 0; j < TN; j++) {
      int n = tn * 16 * TN + j * 16 + m16;
#pragma unroll
      for (int r = 0; r < 8; r++) {
        int m = tm * 16 * TM + i * 16 + kh * 8 + r;
        if (m >= mLim) continue;
        float v = acc[i][j][r] + bv[j];
        if (EPI == 1) v = tanhf(v);
        if (EPI == 2) v = (v > 0.f) ? v : 0.01f * v;
        long off;
        if (transposed) off = (long)b * lC + (long)n * ldc + (m + mOff);
        else            off = (long)b * lC + (long)(m + mOff) * ldc + n;
        if (Ch) Ch[off] = (_Float16)v;
        if (Cf) Cf[off] = v;
      }
    }
  }
}

// ---------------- padding kernel: zero H rows 168..191 -----------------------------------
__global__ void pad_H_kernel(_Float16* __restrict__ H) {
  int b = blockIdx.x;
  for (int i = threadIdx.x; i < 24 * 128; i += 256)
    H[(long)b * (kNP * 128) + (long)(168 + (i >> 7)) * 128 + (i & 127)] = (_Float16)0.f;
}

// ---------------- attention softmax: scores (192,192 f32) -> attn (192,192 f16) ----------
__global__ __launch_bounds__(64) void softmax_kernel(const float* __restrict__ SC,
                                                     _Float16* __restrict__ ATT)
{
  int b = blockIdx.x;
  int n = blockIdx.y;
  int t = threadIdx.x;              // 64 threads
  __shared__ float buf[64];
  const float scale = 0.0883883476483184405f; // 1/sqrt(128)
  const float* row = SC + ((long)b * kNP + n) * kNP;
  _Float16* out = ATT + ((long)b * kNP + n) * kNP;
  if (n >= 168) {
    for (int j = t; j < kNP; j += 64) out[j] = (_Float16)0.f;
    return;
  }
  float mx = -3.402823466e38f;
  for (int j = t; j < 168; j += 64) mx = fmaxf(mx, row[j] * scale);
  buf[t] = mx; __syncthreads();
  for (int o = 32; o > 0; o >>= 1) { if (t < o) buf[t] = fmaxf(buf[t], buf[t + o]); __syncthreads(); }
  mx = buf[0]; __syncthreads();
  float se = 0.f;
  for (int j = t; j < 168; j += 64) se += __expf(row[j] * scale - mx);
  buf[t] = se; __syncthreads();
  for (int o = 32; o > 0; o >>= 1) { if (t < o) buf[t] += buf[t + o]; __syncthreads(); }
  se = buf[0]; __syncthreads();
  float inv = 1.0f / se;
  for (int j = t; j < kNP; j += 64) {
    float v = (j < 168) ? __expf(row[j] * scale - mx) * inv : 0.f;
    out[j] = (_Float16)v;
  }
}

// ---------------- final FC: out[b] (+)= sum_{n<168,e} Oh[b,n,e]*fc_w[n*512+h*128+e] ------
__global__ __launch_bounds__(256) void fc_kernel(const float* __restrict__ Oh,
                                                 const float* __restrict__ fcw,
                                                 const float* __restrict__ fcb,
                                                 float* __restrict__ out,
                                                 int head, int first)
{
  int b = blockIdx.x;
  int t = threadIdx.x;
  __shared__ float red[256];
  float acc = 0.f;
  for (int i = t; i < 168 * 128; i += 256) {
    int n = i >> 7, e = i & 127;
    acc += Oh[((long)b * kNP + n) * 128 + e] * fcw[n * 512 + head * 128 + e];
  }
  red[t] = acc; __syncthreads();
  for (int o = 128; o > 0; o >>= 1) { if (t < o) red[t] += red[t + o]; __syncthreads(); }
  if (t == 0) {
    if (first) out[b] = fcb[0] + red[0];
    else       out[b] += red[0];
  }
}

// =========================================================================================
extern "C" void kernel_launch(void* const* d_in, const int* in_sizes, int n_in,
                              void* d_out, int out_size, void* d_ws, size_t ws_size,
                              hipStream_t stream) {
  const float* X      = (const float*)d_in[0];
  const float* spa_w1 = (const float*)d_in[1];
  const float* spa_b1 = (const float*)d_in[2];
  const float* spa_w2 = (const float*)d_in[3];
  const float* spa_b2 = (const float*)d_in[4];
  const float* tem_w1 = (const float*)d_in[5];
  const float* tem_b1 = (const float*)d_in[6];
  const float* tem_w2 = (const float*)d_in[7];
  const float* tem_b2 = (const float*)d_in[8];
  const float* sgnn_w = (const float*)d_in[9];
  const float* sgnn_b = (const float*)d_in[10];
  const float* tgnn_w = (const float*)d_in[11];
  const float* tgnn_b = (const float*)d_in[12];
  const float* q_w    = (const float*)d_in[13];
  const float* q_b    = (const float*)d_in[14];
  const float* k_w    = (const float*)d_in[15];
  const float* k_b    = (const float*)d_in[16];
  const float* v_w    = (const float*)d_in[17];
  const float* v_b    = (const float*)d_in[18];
  const float* fc_w   = (const float*)d_in[19];
  const float* fc_b   = (const float*)d_in[20];
  float* out = (float*)d_out;

  // ---- workspace carve-up (256B aligned) ----
  char* base = (char*)d_ws;
  size_t off = 0;
  auto alloc = [&](size_t bytes) -> char* {
    char* p = base + off;
    off += (bytes + 255) & ~(size_t)255;
    return p;
  };
  _Float16* W    = (_Float16*)alloc(245760 * 2);                     // weights f16
  float*    FEAT = (float*)   alloc((size_t)kBS * 128 * 20 * 4);     // feats
  _Float16* XFB  = (_Float16*)alloc((size_t)kBS * 128 * 64 * 2);     // Xf  (128,64)
  _Float16* XTB  = (_Float16*)alloc((size_t)kBS * 64 * 128 * 2);     // Xt  (64,128)
  _Float16* T1   = (_Float16*)alloc((size_t)kBS * 48 * 64 * 2);
  _Float16* AS   = (_Float16*)alloc((size_t)kBS * 48 * 64 * 2);
  _Float16* S1   = (_Float16*)alloc((size_t)kBS * 48 * 128 * 2);
  _Float16* T2   = (_Float16*)alloc((size_t)kBS * 128 * 64 * 2);
  _Float16* AT   = (_Float16*)alloc((size_t)kBS * 128 * 128 * 2);
  _Float16* S2   = (_Float16*)alloc((size_t)kBS * 128 * 64 * 2);
  _Float16* H    = (_Float16*)alloc((size_t)kBS * kNP * 128 * 2);    // concat(H_s,H_t) padded
  _Float16* Q    = (_Float16*)alloc((size_t)kBS * kNP * 128 * 2);
  _Float16* KT   = (_Float16*)alloc((size_t)kBS * kNP * 128 * 2);
  _Float16* VT   = (_Float16*)alloc((size_t)kBS * 128 * kNP * 2);    // V transposed (128,192)
  float*    SC   = (float*)   alloc((size_t)kBS * kNP * kNP * 4);
  _Float16* ATTb = (_Float16*)alloc((size_t)kBS * kNP * kNP * 2);
  float*    OH   = (float*)   alloc((size_t)kBS * kNP * 128 * 4);
  (void)ws_size; (void)n_in; (void)in_sizes; (void)out_size;

  const int OFF_SPA1 = 0, OFF_SPA2 = 8192, OFF_TEM1 = 12288, OFF_TEM2 = 16384;
  const int OFF_SGNN = 24576, OFF_TGNN = 40960, OFF_QW = 49152, OFF_KW = 114688, OFF_VW = 180224;

  auto tblk = [](int M, int N, int TM, int TN) {
    return ((M / (16 * TM)) * (N / (16 * TN)) + 3) / 4;
  };
  const long lH = (long)kNP * 128;     // 24576
  const long lS = (long)kNP * kNP;     // 36864

  // 1) weights -> f16 transposed
  prep_weights_kernel<<<240, 256, 0, stream>>>(spa_w1, spa_w2, tem_w1, tem_w2,
                                               sgnn_w, tgnn_w, q_w, k_w, v_w, W);
  // 2) per-signal features
  feat_kernel<<<kBS * 128, 128, 0, stream>>>(X, FEAT);
  // 3) cumsum + normalize -> Xf/Xt (f16, zero padded)
  cumsum_norm_kernel<<<kBS, 128, 0, stream>>>(FEAT, XFB, XTB);

  // 4) T1 = tanh(Xt @ spa_w1 + b1)           (48,64) K=128
  gemm_kernel<1, 128, 1, 2><<<dim3(kBS, tblk(48, 64, 1, 2)), 128, 0, stream>>>(
      XTB, 64 * 128, W + OFF_SPA1, 0, spa_b1, 64, T1, nullptr, 48 * 64, 48, 64, 64, 0, 48, 0);
  // 5) A_s = T1 @ spa_w2 + b2                (48,64) K=64  (cols>=40 zero)
  gemm_kernel<0, 64, 1, 2><<<dim3(kBS, tblk(48, 64, 1, 2)), 128, 0, stream>>>(
      T1, 48 * 64, W + OFF_SPA2, 0, spa_b2, 40, AS, nullptr, 48 * 64, 48, 64, 64, 0, 48, 0);
  // 6) S1 = A_s @ Xt                         (48,128) K=64 ; Bt = Xf (128,64) batched
  gemm_kernel<0, 64, 1, 2><<<dim3(kBS, tblk(48, 128, 1, 2)), 128, 0, stream>>>(
      AS, 48 * 64, XFB, 128 * 64, nullptr, 0, S1, nullptr, 48 * 128, 48, 128, 128, 0, 48, 0);
  // 7) H_s = leaky(S1 @ sgnn_w + b)  -> H rows 0..39
  gemm_kernel<2, 128, 1, 2><<<dim3(kBS, tblk(48, 128, 1, 2)), 128, 0, stream>>>(
      S1, 48 * 128, W + OFF_SGNN, 0, sgnn_b, 128, H, nullptr, lH, 48, 128, 128, 0, 40, 0);
  // 8) T2 = tanh(Xf @ tem_w1 + b1)           (128,64) K=64
  gemm_kernel<1, 64, 2, 2><<<dim3(kBS, tblk(128, 64, 2, 2)), 128, 0, stream>>>(
      XFB, 128 * 64, W + OFF_TEM1, 0, tem_b1, 64, T2, nullptr, 128 * 64, 128, 64, 64, 0, 128, 0);
  // 9) A_t = T2 @ tem_w2 + b2                (128,128) K=64
  gemm_kernel<0, 64, 2, 2><<<dim3(kBS, tblk(128, 128, 2, 2)), 128, 0, stream>>>(
      T2, 128 * 64, W + OFF_TEM2, 0, tem_b2, 128, AT, nullptr, 128 * 128, 128, 128, 128, 0, 128, 0);
  // 10) S2 = A_t @ Xf                        (128,64) K=128 ; Bt = Xt (64,128) batched
  gemm_kernel<0, 128, 2, 2><<<dim3(kBS, tblk(128, 64, 2, 2)), 128, 0, stream>>>(
      AT, 128 * 128, XTB, 64 * 128, nullptr, 0, S2, nullptr, 128 * 64, 128, 64, 64, 0, 128, 0);
  // 11) H_t = leaky(S2 @ tgnn_w + b) -> H rows 40..167
  gemm_kernel<2, 64, 2, 2><<<dim3(kBS, tblk(128, 128, 2, 2)), 128, 0, stream>>>(
      S2, 128 * 64, W + OFF_TGNN, 0, tgnn_b, 128, H, nullptr, lH, 128, 128, 128, 40, 128, 0);
  // 12) zero H pad rows 168..191
  pad_H_kernel<<<kBS, 256, 0, stream>>>(H);

  // 13) attention, head by head
  for (int h = 0; h < 4; h++) {
    // Q = H @ q_w[h] + q_b[h]               (192,128) K=128
    gemm_kernel<0, 128, 2, 2><<<dim3(kBS, tblk(kNP, 128, 2, 2)), 128, 0, stream>>>(
        H, lH, W + OFF_QW + h * 16384, 0, q_b + h * 128, 128,
        Q, nullptr, lH, kNP, 128, 128, 0, kNP, 0);
    // K = H @ k_w[h] + k_b[h]  (row-major == Bt for scores)
    gemm_kernel<0, 128, 2, 2><<<dim3(kBS, tblk(kNP, 128, 2, 2)), 128, 0, stream>>>(
        H, lH, W + OFF_KW + h * 16384, 0, k_b + h * 128, 128,
        KT, nullptr, lH, kNP, 128, 128, 0, kNP, 0);
    // Vt = (H @ v_w[h] + v_b[h])^T          store transposed (128,192)
    gemm_kernel<0, 128, 2, 2><<<dim3(kBS, tblk(kNP, 128, 2, 2)), 128, 0, stream>>>(
        H, lH, W + OFF_VW + h * 16384, 0, v_b + h * 128, 128,
        VT, nullptr, lH, kNP, 128, kNP, 0, kNP, 1);
    // scores = Q @ K^T  (f32)               (192,192) K=128
    gemm_kernel<0, 128, 2, 2><<<dim3(kBS, tblk(kNP, kNP, 2, 2)), 128, 0, stream>>>(
        Q, lH, KT, lH, nullptr, 0,
        nullptr, SC, lS, kNP, kNP, kNP, 0, kNP, 0);
    // softmax (with 1/sqrt(HG) scale) -> attn f16 (192,192); rows/cols >=168 zeroed
    softmax_kernel<<<dim3(kBS, kNP), 64, 0, stream>>>(SC, ATTb);
    // O = attn @ V                          (192,128) K=192 ; Bt = Vt (128,192)
    gemm_kernel<0, 192, 2, 2><<<dim3(kBS, tblk(kNP, 128, 2, 2)), 128, 0, stream>>>(
        ATTb, lS, VT, lH, nullptr, 0,
        nullptr, OH, lH, kNP, 128, 128, 0, kNP, 0);
    // partial FC accumulation into out
    fc_kernel<<<kBS, 256, 0, stream>>>(OH, fc_w, fc_b, out, h, (h == 0) ? 1 : 0);
  }
}
